// MultiHopReasoner_18262200943062
// MI455X (gfx1250) — matrix operand order repaired and checked by reading, not compiled, not run on hardware
//
#include <hip/hip_runtime.h>
#include <hip/hip_bf16.h>

// ---- problem constants --------------------------------------------------
#define Hh      512     // hidden
#define G4      2048    // 4*H (LSTM gates)
#define SEQ     128     // timesteps
#define NCHAIN  8       // 2 query dirs + 3 hops * 2 dirs
#define NTILES  128     // 2048 / 16 output tiles
#define KTILES  16      // 512 / 32 K-chunks
#define NENT    256
#define VOCAB   32000

typedef __attribute__((ext_vector_type(16))) __bf16 v16bf;
typedef __attribute__((ext_vector_type(8)))  float  v8f;

__device__ __forceinline__ unsigned short f32_bf16(float f) {
    union { float f; unsigned u; } x; x.f = f;
    unsigned r = x.u + 0x7FFFu + ((x.u >> 16) & 1u);   // round-to-nearest-even
    return (unsigned short)(r >> 16);
}
__device__ __forceinline__ float sigm(float x) { return 1.0f / (1.0f + expf(-x)); }

// ---- kernel 1: pack Whh (8 chains) into bf16 WMMA-B tiles ---------------
// B tile (32x16, K x N) layout used here (contiguous k, same as A): lanes 0-15 hold K=0..15
// (VGPR v = K{2v,2v+1}), lanes 16-31 hold K=16..31, lane%16 = column N.
// Tile stored lane-major: 32 lanes x 8 dwords = 1KB, so the consumer loads
// 32B/lane contiguously (2x b128).
__global__ void __launch_bounds__(256) pack_whh(const float* __restrict__ qWhh,
                                                const float* __restrict__ eWhh,
                                                unsigned* __restrict__ packB) {
    int wg   = blockIdx.x * 8 + (threadIdx.x >> 5);   // global wave = tile id
    int lane = threadIdx.x & 31;
    int c    = wg >> 11;          // / (KTILES*NTILES)
    int rem  = wg & 2047;
    int kt   = rem >> 7;
    int nt   = rem & 127;
    const float* W = (c < 2) ? (qWhh + (size_t)c * G4 * Hh)
                             : (eWhh + (size_t)(c - 2) * G4 * Hh);
    int n     = nt * 16 + (lane & 15);
    int kbase = kt * 32 + ((lane & 16) ? 16 : 0);
    const float* row = W + (size_t)n * Hh;
    unsigned* out = packB + ((size_t)wg * 256 + lane * 8);
#pragma unroll
    for (int v = 0; v < 8; v++) {
        int k = kbase + 2 * v;
        unsigned lo = f32_bf16(row[k]);
        unsigned hi = f32_bf16(row[k + 1]);
        out[v] = lo | (hi << 16);
    }
}

// ---- kernel 2: 8 independent zero-input LSTM chains, WMMA recurrence ----
// One block per chain; 32 waves each own 64 of the 2048 gate outputs
// (4 accumulator tiles/wave keeps VGPR pressure ~90 regs -> no spills).
// Per step: g = bias + h @ Whh^T via v_wmma_f32_16x16x32_bf16 with h in
// row M=0 of the A fragment (lanes 0 and 16 per the 16-bit A layout).
__global__ void __launch_bounds__(1024) lstm_chains(const unsigned* __restrict__ packB,
                                                    const float* __restrict__ q_bih,
                                                    const float* __restrict__ q_bhh,
                                                    const float* __restrict__ e_bih,
                                                    const float* __restrict__ e_bhh,
                                                    float* __restrict__ finals) {
    __shared__ float          bias[G4];
    __shared__ float          gbuf[G4];
    __shared__ unsigned short hbf[Hh];
    __shared__ float          hF[Hh];
    __shared__ float          cst[Hh];

    const int tid  = threadIdx.x;
    const int lane = tid & 31;
    const int w    = tid >> 5;     // wave 0..31
    const int c    = blockIdx.x;   // chain

    const float* bih = (c < 2) ? q_bih + c * G4 : e_bih + (c - 2) * G4;
    const float* bhh = (c < 2) ? q_bhh + c * G4 : e_bhh + (c - 2) * G4;
    for (int i = tid; i < G4; i += 1024) bias[i] = bih[i] + bhh[i];
    if (tid < Hh) { hbf[tid] = 0; hF[tid] = 0.0f; cst[tid] = 0.0f; }
    __syncthreads();

    const unsigned* pb = packB + (size_t)c * KTILES * NTILES * 256;
    const v8f vzero = {0.f, 0.f, 0.f, 0.f, 0.f, 0.f, 0.f, 0.f};

    for (int step = 0; step < SEQ; ++step) {
        v8f acc[4];
#pragma unroll
        for (int i = 0; i < 4; i++) acc[i] = vzero;

        // unroll 1: keep exactly one kt-iteration of B tiles live so the
        // scheduler cannot prefetch itself into scratch spills.
#pragma unroll 1
        for (int kt = 0; kt < KTILES; ++kt) {
            // A fragment: only row M=0 nonzero -> lanes 0 and 16 carry h.
            union { unsigned short s[16]; v16bf v; } A;
#pragma unroll
            for (int i = 0; i < 16; i++) A.s[i] = 0;
            if ((lane & 15) == 0) {
                const int base = (lane & 16) ? 16 : 0;
                const int k0 = kt * 32;
#pragma unroll
                for (int v2 = 0; v2 < 8; v2++) {
                    int koff = 2 * v2 + base;
                    A.s[2 * v2]     = hbf[k0 + koff];
                    A.s[2 * v2 + 1] = hbf[k0 + koff + 1];
                }
            }
            const unsigned* tb = pb + ((size_t)kt * NTILES + w * 4) * 256 + lane * 8;
#pragma unroll
            for (int ntl = 0; ntl < 4; ++ntl) {
                union { unsigned u[8]; v16bf v; } B;
#pragma unroll
                for (int v2 = 0; v2 < 8; v2++) B.u[v2] = tb[ntl * 256 + v2];
                acc[ntl] = __builtin_amdgcn_wmma_f32_16x16x32_bf16(
                    false, A.v, false, B.v, (short)0, acc[ntl], false, false);
            }
        }
        // D row M=0 = element 0 of the accumulator on lanes 0..15, N = lane.
        if (lane < 16) {
#pragma unroll
            for (int ntl = 0; ntl < 4; ++ntl) {
                int n = w * 64 + ntl * 16 + lane;
                gbuf[n] = acc[ntl][0] + bias[n];
            }
        }
        __syncthreads();
        if (tid < Hh) {   // torch gate order i, f, g, o
            float ig = gbuf[tid], fg = gbuf[Hh + tid];
            float gg = gbuf[2 * Hh + tid], og = gbuf[3 * Hh + tid];
            float cc = sigm(fg) * cst[tid] + sigm(ig) * tanhf(gg);
            float hh = sigm(og) * tanhf(cc);
            cst[tid] = cc; hF[tid] = hh; hbf[tid] = f32_bf16(hh);
        }
        __syncthreads();
    }
    if (tid < Hh) finals[c * Hh + tid] = hF[tid];
}

// ---- kernel 3: 3 GRU hops + uniform attention + answer layer 1 ----------
// Softmax over N bitwise-identical scores is exactly 1/256, so attn is
// written directly and sim_W/sim_b never touch the outputs.
__global__ void __launch_bounds__(1024) gru_answer(const float* __restrict__ finals,
                                                   const float* __restrict__ Wih,
                                                   const float* __restrict__ Whh,
                                                   const float* __restrict__ bih,
                                                   const float* __restrict__ bhh,
                                                   const float* __restrict__ W1,
                                                   const float* __restrict__ b1,
                                                   float* __restrict__ avec,
                                                   float* __restrict__ out_attn) {
    __shared__ float mem[2 * Hh];
    __shared__ float repr[2 * Hh];
    const int t = threadIdx.x;
    mem[t] = finals[t];                       // [query fwd | query bwd]
    __syncthreads();
    for (int hop = 0; hop < 3; ++hop) {
        repr[t] = finals[(2 + 2 * hop) * Hh + t];   // [ent fwd | ent bwd]
        __syncthreads();
        float gi[3], gh[3];
#pragma unroll
        for (int g = 0; g < 3; g++) {               // torch order r, z, n
            const float* wi = Wih + (size_t)(g * 1024 + t) * 1024;
            const float* wh = Whh + (size_t)(g * 1024 + t) * 1024;
            float si = 0.f, sh = 0.f;
            for (int k = 0; k < 1024; k++) { si += wi[k] * repr[k]; sh += wh[k] * mem[k]; }
            gi[g] = si + bih[g * 1024 + t];
            gh[g] = sh + bhh[g * 1024 + t];
        }
        float r = sigm(gi[0] + gh[0]);
        float z = sigm(gi[1] + gh[1]);
        float n = tanhf(gi[2] + r * gh[2]);
        float newm = (1.0f - z) * n + z * mem[t];
        __syncthreads();
        mem[t] = newm;
        __syncthreads();
    }
    if (t < 3 * NENT) out_attn[t] = 1.0f / 256.0f;   // exact uniform softmax
    if (t < Hh) {
        const float* w1 = W1 + (size_t)t * 1024;
        float s = 0.f;
        for (int k = 0; k < 1024; k++) s += w1[k] * mem[k];
        s += b1[t];
        avec[t] = s > 0.f ? s : 0.f;                 // ReLU
    }
}

// ---- kernel 4: 32000x512 vocab matvec (HBM bandwidth bound) -------------
// One wave per row, fully coalesced float4 loads, wave32 shuffle reduction.
__global__ void __launch_bounds__(256) vocab_logits(const float* __restrict__ avec,
                                                    const float* __restrict__ W2,
                                                    const float* __restrict__ b2,
                                                    float* __restrict__ out) {
    __shared__ __align__(16) float aL[Hh];
    const int t = threadIdx.x;
    aL[t] = avec[t];
    aL[t + 256] = avec[t + 256];
    __syncthreads();
    const int lane = t & 31;
    const int w    = t >> 5;
    const int v    = blockIdx.x * 8 + w;
    if (v + 8 < VOCAB)   // hint next row group into GL2 (global_prefetch_b8)
        __builtin_prefetch(W2 + (size_t)(v + 8) * Hh + lane * 16, 0, 1);
    const float4* Wr = (const float4*)(W2 + (size_t)v * Hh);
    const float4* aV = (const float4*)aL;
    float s = 0.f;
#pragma unroll
    for (int j = 0; j < 4; j++) {
        float4 wv = Wr[j * 32 + lane];
        float4 av = aV[j * 32 + lane];
        s += wv.x * av.x + wv.y * av.y + wv.z * av.z + wv.w * av.w;
    }
#pragma unroll
    for (int off = 16; off; off >>= 1) s += __shfl_xor(s, off, 32);
    if (lane == 0) out[v] = s + b2[v];
}

// ---- launcher -----------------------------------------------------------
extern "C" void kernel_launch(void* const* d_in, const int* in_sizes, int n_in,
                              void* d_out, int out_size, void* d_ws, size_t ws_size,
                              hipStream_t stream) {
    const float* qWhh = (const float*)d_in[1];
    const float* qbih = (const float*)d_in[2];
    const float* qbhh = (const float*)d_in[3];
    const float* eWhh = (const float*)d_in[5];
    const float* ebih = (const float*)d_in[6];
    const float* ebhh = (const float*)d_in[7];
    // d_in[0],[4]  (W_ih): input is identically zero -> unused
    // d_in[8],[9]  (sim_W/sim_b): softmax of equal scores is uniform -> unused
    // d_in[18..20] (ids, num_entities): zero-embedding quirk -> unused
    const float* gWih = (const float*)d_in[10];
    const float* gWhh = (const float*)d_in[11];
    const float* gbih = (const float*)d_in[12];
    const float* gbhh = (const float*)d_in[13];
    const float* aW1  = (const float*)d_in[14];
    const float* ab1  = (const float*)d_in[15];
    const float* aW2  = (const float*)d_in[16];
    const float* ab2  = (const float*)d_in[17];
    float* out = (float*)d_out;

    // workspace: [ packed bf16 B tiles : 16 MB ][ finals : 8*512 f32 ][ avec : 512 f32 ]
    unsigned* packB = (unsigned*)d_ws;
    const size_t PACK_DW = (size_t)NCHAIN * KTILES * NTILES * 256;
    float* finals = (float*)((char*)d_ws + PACK_DW * 4);
    float* avec   = finals + NCHAIN * Hh;

    pack_whh    <<<dim3(2048),      dim3(256),  0, stream>>>(qWhh, eWhh, packB);
    lstm_chains <<<dim3(NCHAIN),    dim3(1024), 0, stream>>>(packB, qbih, qbhh, ebih, ebhh, finals);
    gru_answer  <<<dim3(1),         dim3(1024), 0, stream>>>(finals, gWih, gWhh, gbih, gbhh,
                                                             aW1, ab1, avec, out);
    vocab_logits<<<dim3(VOCAB / 8), dim3(256),  0, stream>>>(avec, aW2, ab2, out + 3 * NENT);

    (void)in_sizes; (void)n_in; (void)out_size; (void)ws_size;
}
